// SuperpixelConv_68633577390527
// MI455X (gfx1250) — matrix operand order repaired and checked by reading, not compiled, not run on hardware
//
#include <hip/hip_runtime.h>
#include <hip/hip_bf16.h>

typedef __attribute__((ext_vector_type(16))) __bf16 v16bf;
typedef __attribute__((ext_vector_type(8)))  __bf16 v8bf;
typedef __attribute__((ext_vector_type(8)))  float  v8f;
typedef __attribute__((ext_vector_type(4)))  float  v4f;

#define B_   2
#define C_   96
#define H_   128
#define W_   128
#define F_   9
#define KS   5
#define K2   25
#define O_   96
#define P_   (H_*W_)

#define TPX    64           // pixels per workgroup (row strip)
#define HALO_W (TPX + 4)    // 68
#define CPAD   104          // channel pad: 208B row stride -> 16B aligned, conflict-free

// workspace layout (bytes)
#define BWS_BYTES  (O_*K2*C_*2)           // bf16 weights [o][k][c]
#define WABS_OFF   (BWS_BYTES)            // f32 [o][k] abs col sums
#define RWS_OFF    (WABS_OFF + O_*K2*4)   // f32 r [b][k][p]

// ---------------- kernel 0: weight prep (bf16 transpose + abs col sums) -------------
__global__ void spx_prep_weights(const float* __restrict__ weight,
                                 __bf16* __restrict__ Bws, float* __restrict__ wabs) {
    int t = blockIdx.x * blockDim.x + threadIdx.x;
    if (t >= O_ * K2) return;
    int o = t / K2, kk = t % K2;
    const float* wrow = weight + o * (C_ * K2) + kk;  // weight[o][c*25+kk]
    __bf16* dst = Bws + (o * K2 + kk) * C_;
    float s = 0.f;
    for (int c = 0; c < C_; ++c) {
        float v = wrow[c * K2];
        s += fabsf(v);
        dst[c] = (__bf16)v;
    }
    wabs[t] = s;
}

// ---------------- kernel 1: similarity reweighting r[b][k][p] -----------------------
__global__ void spx_compute_r(const float* __restrict__ sims, float* __restrict__ rws) {
    int t = blockIdx.x * blockDim.x + threadIdx.x;   // B*P threads
    int b = t / P_;
    int p = t - b * P_;
    int h = p / W_, w = p - h * W_;
    const float* sb = sims + b * F_ * P_;
    float s0[F_];
    #pragma unroll
    for (int f = 0; f < F_; ++f) s0[f] = sb[f * P_ + p];
    float attn[K2];
    float tot = 0.f;
    #pragma unroll
    for (int kk = 0; kk < K2; ++kk) {
        int hh = h + kk / KS - 2;
        int ww = w + kk % KS - 2;
        float a = 0.f;
        if (hh >= 0 && hh < H_ && ww >= 0 && ww < W_) {
            int q = hh * W_ + ww;
            #pragma unroll
            for (int f = 0; f < F_; ++f) a += s0[f] * sb[f * P_ + q];
        }
        attn[kk] = a;
        tot += a;
    }
    float inv = 1.f / (1e-5f + tot);
    #pragma unroll
    for (int kk = 0; kk < K2; ++kk) rws[(b * K2 + kk) * P_ + p] = attn[kk] * inv;
}

// ---------------- kernel 2: main WMMA GEMM ------------------------------------------
__global__ __launch_bounds__(256) void spx_main(
    const float* __restrict__ x, const float* __restrict__ bias,
    const __bf16* __restrict__ Bws, const float* __restrict__ wabs,
    const float* __restrict__ rws, float* __restrict__ out)
{
    __shared__ __bf16 xsh[5 * HALO_W * CPAD];   // 70720 B, layout [row][col][c]
    __shared__ float  rT[K2 * TPX];             // 6400 B,  layout [kk][pixel]

    int blk = blockIdx.x;                // 512 blocks: b(2) x h(128) x wtile(2)
    int b   = blk >> 8;
    int rem = blk & 255;
    int h   = rem >> 1;
    int w0  = (rem & 1) << 6;

    int tid = threadIdx.x;

    // stage x halo (rows h-2..h+2, cols w0-2..w0+65, all 96 ch) as bf16, zero-padded
    for (int idx = tid; idx < C_ * 5 * HALO_W; idx += 256) {
        int c   = idx / (5 * HALO_W);
        int r2  = idx - c * (5 * HALO_W);
        int row = r2 / HALO_W;
        int col = r2 - row * HALO_W;
        int gh = h + row - 2;
        int gw = w0 + col - 2;
        float v = 0.f;
        if (gh >= 0 && gh < H_ && gw >= 0 && gw < W_)
            v = x[((b * C_ + c) * H_ + gh) * W_ + gw];
        xsh[(row * HALO_W + col) * CPAD + c] = (__bf16)v;
    }
    // stage r for this strip
    for (int idx = tid; idx < K2 * TPX; idx += 256) {
        int kk = idx / TPX, j = idx - kk * TPX;
        rT[kk * TPX + j] = rws[(b * K2 + kk) * P_ + h * W_ + w0 + j];
    }
    __syncthreads();

    int lane = tid & 31;
    int wid  = tid >> 5;
    int wr   = wid & 3;      // pixel group: pixels wr*16 .. wr*16+15
    int wc   = wid >> 2;     // output half: o in [wc*48, wc*48+48)
    int n    = lane & 15;
    int kh   = lane >> 4;
    int obase = wc * 48;
    int pc    = wr * 16 + n; // A-row pixel (M = n)

    v8f acc0 = {}; v8f acc1 = {}; v8f acc2 = {};
    float den0[8] = {}, den1[8] = {}, den2[8] = {};

    #pragma unroll 1
    for (int kk = 0; kk < K2; ++kk) {
        const __bf16* xrow = &xsh[((kk / KS) * HALO_W + (pc + kk % KS)) * CPAD];
        const __bf16* brow = Bws + kk * C_ + kh * 16;

        // ---- partial T_kk[p,o] = sum_c x_bf16 * w_bf16 over 3 chunks of 32 ----
        v8f p0, p1, p2;
        #pragma unroll
        for (int cc = 0; cc < 3; ++cc) {
            // A fragment: K(e) = 16*kh + e, c = 32*cc + K(e)
            int c0 = cc * 32 + kh * 16;
            v8bf a0 = *(const v8bf*)(xrow + c0);        // e = 0..7
            v8bf a1 = *(const v8bf*)(xrow + c0 + 8);    // e = 8..15
            v16bf a = __builtin_shufflevector(a0, a1, 0,1,2,3,4,5,6,7,8,9,10,11,12,13,14,15);
            // B fragments: K(e) = 16*kh + e, N = n  ->  16 consecutive c
            v16bf bb0 = *(const v16bf*)(brow + (obase      + n) * (K2 * C_) + cc * 32);
            v16bf bb1 = *(const v16bf*)(brow + (obase + 16 + n) * (K2 * C_) + cc * 32);
            v16bf bb2 = *(const v16bf*)(brow + (obase + 32 + n) * (K2 * C_) + cc * 32);
            if (cc == 0) {
                v8f z = {};
                p0 = __builtin_amdgcn_wmma_f32_16x16x32_bf16(false, a, false, bb0, (short)0, z, false, false);
                p1 = __builtin_amdgcn_wmma_f32_16x16x32_bf16(false, a, false, bb1, (short)0, z, false, false);
                p2 = __builtin_amdgcn_wmma_f32_16x16x32_bf16(false, a, false, bb2, (short)0, z, false, false);
            } else {
                p0 = __builtin_amdgcn_wmma_f32_16x16x32_bf16(false, a, false, bb0, (short)0, p0, false, false);
                p1 = __builtin_amdgcn_wmma_f32_16x16x32_bf16(false, a, false, bb1, (short)0, p1, false, false);
                p2 = __builtin_amdgcn_wmma_f32_16x16x32_bf16(false, a, false, bb2, (short)0, p2, false, false);
            }
        }

        // ---- fold r (f32) into accumulators; D rows: M = i + 8*kh ----
        const float* rp = &rT[kk * TPX + wr * 16 + kh * 8];
        v4f ra = *(const v4f*)rp;
        v4f rb = *(const v4f*)(rp + 4);
        // denominator terms (f32, |r| via fabsf -> free VALU modifier)
        float wa0 = wabs[(obase      + n) * K2 + kk];
        float wa1 = wabs[(obase + 16 + n) * K2 + kk];
        float wa2 = wabs[(obase + 32 + n) * K2 + kk];
        #pragma unroll
        for (int i = 0; i < 4; ++i) {
            acc0[i]     += ra[i] * p0[i];
            acc0[i + 4] += rb[i] * p0[i + 4];
            acc1[i]     += ra[i] * p1[i];
            acc1[i + 4] += rb[i] * p1[i + 4];
            acc2[i]     += ra[i] * p2[i];
            acc2[i + 4] += rb[i] * p2[i + 4];
            den0[i]     += fabsf(ra[i]) * wa0;
            den0[i + 4] += fabsf(rb[i]) * wa0;
            den1[i]     += fabsf(ra[i]) * wa1;
            den1[i + 4] += fabsf(rb[i]) * wa1;
            den2[i]     += fabsf(ra[i]) * wa2;
            den2[i + 4] += fabsf(rb[i]) * wa2;
        }
    }

    int pixbase = h * W_ + w0 + wr * 16 + kh * 8;
    auto writeTile = [&](const v8f& acc, const float* den, int o) {
        float bo = bias[o];
        float* dst = out + (b * O_ + o) * P_ + pixbase;
        v4f lo, hi;
        #pragma unroll
        for (int i = 0; i < 4; ++i) {
            lo[i] = acc[i]     / (1e-10f + den[i])     + bo;
            hi[i] = acc[i + 4] / (1e-10f + den[i + 4]) + bo;
        }
        *(v4f*)dst       = lo;
        *(v4f*)(dst + 4) = hi;
    };
    writeTile(acc0, den0, obase + n);
    writeTile(acc1, den1, obase + 16 + n);
    writeTile(acc2, den2, obase + 32 + n);
}

// ---------------- launcher ----------------------------------------------------------
extern "C" void kernel_launch(void* const* d_in, const int* in_sizes, int n_in,
                              void* d_out, int out_size, void* d_ws, size_t ws_size,
                              hipStream_t stream) {
    const float* x      = (const float*)d_in[0];
    const float* sims   = (const float*)d_in[1];
    const float* weight = (const float*)d_in[2];
    const float* bias   = (const float*)d_in[3];
    float* out = (float*)d_out;

    char* ws   = (char*)d_ws;
    __bf16* Bws = (__bf16*)ws;
    float* wabs = (float*)(ws + WABS_OFF);
    float* rws  = (float*)(ws + RWS_OFF);

    spx_prep_weights<<<(O_ * K2 + 255) / 256, 256, 0, stream>>>(weight, Bws, wabs);
    spx_compute_r<<<(B_ * P_) / 256, 256, 0, stream>>>(sims, rws);
    spx_main<<<B_ * H_ * (W_ / TPX), 256, 0, stream>>>(x, bias, Bws, wabs, rws, out);
}
